// CohereMoe_54245436949032
// MI455X (gfx1250) — matrix operand order repaired and checked by reading, not compiled, not run on hardware
//
#include <hip/hip_runtime.h>
#include <hip/hip_bf16.h>

// ---------------- problem constants ----------------
#define T_   8192
#define H_   1024
#define I_   2048
#define E_   16
#define TOPK_ 2
#define IS_  2048
#define MT   32                    // rows per block tile (bucket alignment)
#define SCAP (T_*TOPK_ + E_*MT)    // 16896: flat assignment capacity, 32-aligned buckets
#define KSTAGE 128                 // K elements staged in LDS per barrier
#define AROWQ 17                   // padded row stride in uint4 (128 bf16 + 8 pad)

typedef __attribute__((ext_vector_type(16))) __bf16 v16bf;
typedef __attribute__((ext_vector_type(8)))  float  v8f;

union Frag16 { v16bf v; uint4 q[2]; };   // 32B: one WMMA bf16 A/B fragment

// ---------------- small utility kernels ----------------
__global__ void zero_i32(int* __restrict__ p, int n) {
    int i = blockIdx.x * 256 + threadIdx.x;
    if (i < n) p[i] = 0;
}

// fp32 -> bf16 elementwise (n multiple of 4)
__global__ void cvt_bf16(const float* __restrict__ s, __hip_bfloat16* __restrict__ d, size_t n) {
    size_t base = ((size_t)blockIdx.x * 256 + threadIdx.x) * 4;
    if (base >= n) return;
    float4 v = *(const float4*)(s + base);
    d[base + 0] = __float2bfloat16(v.x);
    d[base + 1] = __float2bfloat16(v.y);
    d[base + 2] = __float2bfloat16(v.z);
    d[base + 3] = __float2bfloat16(v.w);
}

// per-slice tiled transpose + convert: src [R][C] f32 -> dst [C][R] bf16
__global__ void __launch_bounds__(256)
transpose_cvt(const float* __restrict__ src, __hip_bfloat16* __restrict__ dst, int R, int C) {
    __shared__ float tile[32][33];
    size_t off = (size_t)blockIdx.z * R * C;
    int r0 = blockIdx.y * 32, c0 = blockIdx.x * 32;
    #pragma unroll
    for (int i = threadIdx.y; i < 32; i += 8)
        tile[i][threadIdx.x] = src[off + (size_t)(r0 + i) * C + c0 + threadIdx.x];
    __syncthreads();
    #pragma unroll
    for (int i = threadIdx.y; i < 32; i += 8)
        dst[off + (size_t)(c0 + i) * R + r0 + threadIdx.x] = __float2bfloat16(tile[threadIdx.x][i]);
}

// ---------------- router: logits, sigmoid, top-2, renorm, counts ----------------
__global__ void __launch_bounds__(256)
router_k(const float* __restrict__ x, const float* __restrict__ gw,
         int* __restrict__ topk_id, float* __restrict__ topk_w, int* __restrict__ counts) {
    int t = blockIdx.x * 256 + threadIdx.x;
    const float* xr = x + (size_t)t * H_;
    float acc[E_];
    #pragma unroll
    for (int e = 0; e < E_; ++e) acc[e] = 0.f;
    for (int h = 0; h < H_; ++h) {
        float xv = xr[h];
        const float* g = gw + h * E_;
        #pragma unroll
        for (int e = 0; e < E_; ++e) acc[e] = fmaf(xv, g[e], acc[e]);
    }
    float b0 = -1.f, b1 = -1.f; int i0 = 0, i1 = 0;
    #pragma unroll
    for (int e = 0; e < E_; ++e) {
        float s = 1.f / (1.f + __expf(-acc[e]));
        if (s > b0)      { b1 = b0; i1 = i0; b0 = s; i0 = e; }
        else if (s > b1) { b1 = s;  i1 = e; }
    }
    float inv = 1.f / (b0 + b1);
    topk_id[2*t] = i0;  topk_id[2*t+1] = i1;
    topk_w[2*t] = b0 * inv;  topk_w[2*t+1] = b1 * inv;
    atomicAdd(&counts[i0], 1);
    atomicAdd(&counts[i1], 1);
}

// MT-aligned exclusive prefix of bucket sizes; zero fill counters
__global__ void offsets_k(const int* __restrict__ counts, int* __restrict__ off, int* __restrict__ fill) {
    if (threadIdx.x == 0) {
        int a = 0;
        for (int e = 0; e < E_; ++e) {
            off[e] = a;
            a += (counts[e] + (MT - 1)) & ~(MT - 1);
            fill[e] = 0;
        }
        off[E_] = a;   // total padded rows
    }
}

__global__ void __launch_bounds__(256)
scatter_k(const int* __restrict__ topk_id, int* __restrict__ off, int* __restrict__ fill,
          int* __restrict__ assign_tok, int* __restrict__ assign_e, int* __restrict__ tok_pos) {
    int t = blockIdx.x * 256 + threadIdx.x;
    #pragma unroll
    for (int k = 0; k < TOPK_; ++k) {
        int e = topk_id[2*t + k];
        int pos = off[e] + atomicAdd(&fill[e], 1);
        assign_tok[pos] = t;
        assign_e[pos]   = e;
        tok_pos[2*t + k] = pos;
    }
}

// ---------------- GEMM1: act = silu(Arows @ Bt_g) * (Arows @ Bt_u), bf16 out ----------------
// Block tile: 32 rows x 256 cols (of Iout). A: bf16 [*, Kd]; Bt: bf16 [2*Iout, Kd] per expert.
__global__ void __launch_bounds__(256)
gemm1_silu(const __hip_bfloat16* __restrict__ A, const __hip_bfloat16* __restrict__ Bt,
           __hip_bfloat16* __restrict__ out,
           const int* __restrict__ rowTok, const int* __restrict__ rowExp,
           const int* __restrict__ rowLimit, int Kd, int Iout) {
    const int rt = blockIdx.x;
    if (rowLimit && rt * MT >= *rowLimit) return;
    const int tid = threadIdx.x;
    const int lane = tid & 31, wave = tid >> 5;

    __shared__ int ldsTok[MT];
    __shared__ __align__(16) uint4 ldsA[2][MT * AROWQ];   // double-buffered 32 x 128 bf16, padded rows

    if (tid < MT)
        ldsTok[tid] = rowTok ? rowTok[rt * MT + tid] : (rt * MT + tid);
    __syncthreads();

    const __hip_bfloat16* Be = Bt;
    if (rowExp) Be += (size_t)rowExp[rt * MT] * ((size_t)2 * Iout * Kd);

    const int m = lane & 15, half = lane >> 4;
    const int ng = blockIdx.y * 256 + wave * 32;      // gate-column base for this wave
    const size_t bn_g0 = (size_t)(ng + m) * Kd;
    const size_t bn_g1 = (size_t)(ng + 16 + m) * Kd;
    const size_t bn_u0 = (size_t)(ng + Iout + m) * Kd;
    const size_t bn_u1 = (size_t)(ng + Iout + 16 + m) * Kd;
    const int kh = 8 * half;

    v8f cg0a = {}, cg1a = {}, cu0a = {}, cu1a = {};   // rows 0-15
    v8f cg0b = {}, cg1b = {}, cu0b = {}, cu1b = {};   // rows 16-31

    const int ar0 = tid >> 4;         // 0..15 (also handles row ar0+16)
    const int achk = tid & 15;        // 16B chunk within 128-elem row
    const uint4* asrc0 = (const uint4*)(A + (size_t)ldsTok[ar0] * Kd) + achk;
    const uint4* asrc1 = (const uint4*)(A + (size_t)ldsTok[ar0 + 16] * Kd) + achk;
    const int ldst0 = ar0 * AROWQ + achk;
    const int ldst1 = (ar0 + 16) * AROWQ + achk;

    const int nstages = Kd / KSTAGE;
    uint4 areg0 = asrc0[0];
    uint4 areg1 = asrc1[0];
    for (int s = 0; s < nstages; ++s) {
        ldsA[s & 1][ldst0] = areg0;
        ldsA[s & 1][ldst1] = areg1;
        __syncthreads();
        if (s + 1 < nstages) {
            areg0 = asrc0[(s + 1) * (KSTAGE / 8)];
            areg1 = asrc1[(s + 1) * (KSTAGE / 8)];
        }
        const uint4* ap = &ldsA[s & 1][0];
        const int kb = s * KSTAGE;
        #pragma unroll
        for (int ks = 0; ks < KSTAGE / 32; ++ks) {
            Frag16 a0, a1, bg0, bg1, bu0, bu1;
            a0.q[0] = ap[m * AROWQ + 4 * ks + half];
            a0.q[1] = ap[m * AROWQ + 4 * ks + 2 + half];
            a1.q[0] = ap[(16 + m) * AROWQ + 4 * ks + half];
            a1.q[1] = ap[(16 + m) * AROWQ + 4 * ks + 2 + half];
            const int ko = kb + 32 * ks + kh;
            const uint4* bp;
            bp = (const uint4*)(Be + bn_g0 + ko); bg0.q[0] = bp[0]; bg0.q[1] = bp[2];
            bp = (const uint4*)(Be + bn_g1 + ko); bg1.q[0] = bp[0]; bg1.q[1] = bp[2];
            bp = (const uint4*)(Be + bn_u0 + ko); bu0.q[0] = bp[0]; bu0.q[1] = bp[2];
            bp = (const uint4*)(Be + bn_u1 + ko); bu1.q[0] = bp[0]; bu1.q[1] = bp[2];
            cg0a = __builtin_amdgcn_wmma_f32_16x16x32_bf16(false, a0.v, false, bg0.v, (short)0, cg0a, false, false);
            cg1a = __builtin_amdgcn_wmma_f32_16x16x32_bf16(false, a0.v, false, bg1.v, (short)0, cg1a, false, false);
            cu0a = __builtin_amdgcn_wmma_f32_16x16x32_bf16(false, a0.v, false, bu0.v, (short)0, cu0a, false, false);
            cu1a = __builtin_amdgcn_wmma_f32_16x16x32_bf16(false, a0.v, false, bu1.v, (short)0, cu1a, false, false);
            cg0b = __builtin_amdgcn_wmma_f32_16x16x32_bf16(false, a1.v, false, bg0.v, (short)0, cg0b, false, false);
            cg1b = __builtin_amdgcn_wmma_f32_16x16x32_bf16(false, a1.v, false, bg1.v, (short)0, cg1b, false, false);
            cu0b = __builtin_amdgcn_wmma_f32_16x16x32_bf16(false, a1.v, false, bu0.v, (short)0, cu0b, false, false);
            cu1b = __builtin_amdgcn_wmma_f32_16x16x32_bf16(false, a1.v, false, bu1.v, (short)0, cu1b, false, false);
        }
    }

    // D layout: lane holds column n = lane&15, rows m = 8*(lane>>4)+r
    const int nloc = m;
    const int mbase = 8 * half;
    #pragma unroll
    for (int r = 0; r < 8; ++r) {
        size_t rowA = (size_t)rt * MT + mbase + r;        // rows 0-15 subtile
        size_t rowB = rowA + 16;                          // rows 16-31 subtile
        float g, u;
        g = cg0a[r]; u = cu0a[r];
        out[rowA * Iout + ng + nloc]      = __float2bfloat16((g / (1.f + __expf(-g))) * u);
        g = cg1a[r]; u = cu1a[r];
        out[rowA * Iout + ng + 16 + nloc] = __float2bfloat16((g / (1.f + __expf(-g))) * u);
        g = cg0b[r]; u = cu0b[r];
        out[rowB * Iout + ng + nloc]      = __float2bfloat16((g / (1.f + __expf(-g))) * u);
        g = cg1b[r]; u = cu1b[r];
        out[rowB * Iout + ng + 16 + nloc] = __float2bfloat16((g / (1.f + __expf(-g))) * u);
    }
}

// ---------------- GEMM2: out(f32) = A(bf16 [*,Kd]) @ Bt(bf16 [Nd,Kd]) ----------------
// Block tile: 32 rows x 256 cols of Nd.
__global__ void __launch_bounds__(256)
gemm2_k(const __hip_bfloat16* __restrict__ A, const __hip_bfloat16* __restrict__ Bt,
        float* __restrict__ out, const int* __restrict__ rowExp,
        const int* __restrict__ rowLimit, int Kd, int Nd) {
    const int rt = blockIdx.x;
    if (rowLimit && rt * MT >= *rowLimit) return;
    const int tid = threadIdx.x;
    const int lane = tid & 31, wave = tid >> 5;

    __shared__ __align__(16) uint4 ldsA[2][MT * AROWQ];

    const __hip_bfloat16* Be = Bt;
    if (rowExp) Be += (size_t)rowExp[rt * MT] * ((size_t)Nd * Kd);

    const int m = lane & 15, half = lane >> 4;
    const int nb = blockIdx.y * 256 + wave * 32;
    const size_t bn0 = (size_t)(nb + m) * Kd;
    const size_t bn1 = (size_t)(nb + 16 + m) * Kd;
    const int kh = 8 * half;

    v8f c0a = {}, c1a = {}, c0b = {}, c1b = {};

    const int ar0 = tid >> 4, achk = tid & 15;
    const uint4* asrc0 = (const uint4*)(A + (size_t)(rt * MT + ar0) * Kd) + achk;
    const uint4* asrc1 = (const uint4*)(A + (size_t)(rt * MT + ar0 + 16) * Kd) + achk;
    const int ldst0 = ar0 * AROWQ + achk;
    const int ldst1 = (ar0 + 16) * AROWQ + achk;

    const int nstages = Kd / KSTAGE;
    uint4 areg0 = asrc0[0];
    uint4 areg1 = asrc1[0];
    for (int s = 0; s < nstages; ++s) {
        ldsA[s & 1][ldst0] = areg0;
        ldsA[s & 1][ldst1] = areg1;
        __syncthreads();
        if (s + 1 < nstages) {
            areg0 = asrc0[(s + 1) * (KSTAGE / 8)];
            areg1 = asrc1[(s + 1) * (KSTAGE / 8)];
        }
        const uint4* ap = &ldsA[s & 1][0];
        const int kb = s * KSTAGE;
        #pragma unroll
        for (int ks = 0; ks < KSTAGE / 32; ++ks) {
            Frag16 a0, a1, b0, b1;
            a0.q[0] = ap[m * AROWQ + 4 * ks + half];
            a0.q[1] = ap[m * AROWQ + 4 * ks + 2 + half];
            a1.q[0] = ap[(16 + m) * AROWQ + 4 * ks + half];
            a1.q[1] = ap[(16 + m) * AROWQ + 4 * ks + 2 + half];
            const int ko = kb + 32 * ks + kh;
            const uint4* bp;
            bp = (const uint4*)(Be + bn0 + ko); b0.q[0] = bp[0]; b0.q[1] = bp[2];
            bp = (const uint4*)(Be + bn1 + ko); b1.q[0] = bp[0]; b1.q[1] = bp[2];
            c0a = __builtin_amdgcn_wmma_f32_16x16x32_bf16(false, a0.v, false, b0.v, (short)0, c0a, false, false);
            c1a = __builtin_amdgcn_wmma_f32_16x16x32_bf16(false, a0.v, false, b1.v, (short)0, c1a, false, false);
            c0b = __builtin_amdgcn_wmma_f32_16x16x32_bf16(false, a1.v, false, b0.v, (short)0, c0b, false, false);
            c1b = __builtin_amdgcn_wmma_f32_16x16x32_bf16(false, a1.v, false, b1.v, (short)0, c1b, false, false);
        }
    }

    const int nloc = m, mbase = 8 * half;
    #pragma unroll
    for (int r = 0; r < 8; ++r) {
        size_t rowA = (size_t)rt * MT + mbase + r;
        size_t rowB = rowA + 16;
        out[rowA * Nd + nb + nloc]      = c0a[r];
        out[rowA * Nd + nb + 16 + nloc] = c1a[r];
        out[rowB * Nd + nb + nloc]      = c0b[r];
        out[rowB * Nd + nb + 16 + nloc] = c1b[r];
    }
}

// ---------------- final combine: out[t] += w0*y[p0] + w1*y[p1] ----------------
__global__ void __launch_bounds__(256)
combine_k(float* __restrict__ out, const float* __restrict__ y,
          const int* __restrict__ tok_pos, const float* __restrict__ topk_w) {
    int t = blockIdx.x;
    int p0 = tok_pos[2*t], p1 = tok_pos[2*t+1];
    float w0 = topk_w[2*t], w1 = topk_w[2*t+1];
    const float4* y0 = (const float4*)(y + (size_t)p0 * H_);
    const float4* y1 = (const float4*)(y + (size_t)p1 * H_);
    float4* o = (float4*)(out + (size_t)t * H_);
    int i = threadIdx.x;                       // 256 * float4 == 1024 floats
    float4 a = o[i], v0 = y0[i], v1 = y1[i];
    a.x += w0 * v0.x + w1 * v1.x;
    a.y += w0 * v0.y + w1 * v1.y;
    a.z += w0 * v0.z + w1 * v1.z;
    a.w += w0 * v0.w + w1 * v1.w;
    o[i] = a;
}

// ---------------- host launch ----------------
extern "C" void kernel_launch(void* const* d_in, const int* in_sizes, int n_in,
                              void* d_out, int out_size, void* d_ws, size_t ws_size,
                              hipStream_t stream) {
    const float* x      = (const float*)d_in[0];
    const float* gate_w = (const float*)d_in[1];
    const float* w1     = (const float*)d_in[2];
    const float* w2     = (const float*)d_in[3];
    const float* ws1    = (const float*)d_in[4];
    const float* ws2    = (const float*)d_in[5];
    float* out = (float*)d_out;

    char* p = (char*)d_ws;
    auto alloc = [&](size_t bytes) -> void* {
        void* r = (void*)p;
        p += (bytes + 255) & ~(size_t)255;
        return r;
    };
    __hip_bfloat16* xb   = (__hip_bfloat16*)alloc((size_t)T_ * H_ * 2);
    __hip_bfloat16* w1t  = (__hip_bfloat16*)alloc((size_t)E_ * 2 * I_ * H_ * 2);
    __hip_bfloat16* w2t  = (__hip_bfloat16*)alloc((size_t)E_ * H_ * I_ * 2);
    __hip_bfloat16* ws1t = (__hip_bfloat16*)alloc((size_t)2 * IS_ * H_ * 2);
    __hip_bfloat16* ws2t = (__hip_bfloat16*)alloc((size_t)H_ * IS_ * 2);
    __hip_bfloat16* act  = (__hip_bfloat16*)alloc((size_t)SCAP * I_ * 2);  // also reused as hs
    float* y             = (float*)alloc((size_t)SCAP * H_ * 4);
    int*   topk_id       = (int*)alloc((size_t)T_ * 2 * 4);
    float* topk_w        = (float*)alloc((size_t)T_ * 2 * 4);
    int*   tok_pos       = (int*)alloc((size_t)T_ * 2 * 4);
    int*   assign_tok    = (int*)alloc((size_t)SCAP * 4);
    int*   assign_e      = (int*)alloc((size_t)SCAP * 4);
    int*   counts        = (int*)alloc(64);
    int*   fill          = (int*)alloc(64);
    int*   off           = (int*)alloc(68);   // 17 ints

    // --- precision conversion: fp32 -> bf16 (weights transposed to [N][K]) ---
    cvt_bf16<<<(T_ * H_ / 4 + 255) / 256, 256, 0, stream>>>(x, xb, (size_t)T_ * H_);
    transpose_cvt<<<dim3(2 * I_ / 32, H_ / 32, E_), dim3(32, 8), 0, stream>>>(w1, w1t, H_, 2 * I_);
    transpose_cvt<<<dim3(H_ / 32, I_ / 32, E_), dim3(32, 8), 0, stream>>>(w2, w2t, I_, H_);
    transpose_cvt<<<dim3(2 * IS_ / 32, H_ / 32, 1), dim3(32, 8), 0, stream>>>(ws1, ws1t, H_, 2 * IS_);
    transpose_cvt<<<dim3(H_ / 32, IS_ / 32, 1), dim3(32, 8), 0, stream>>>(ws2, ws2t, IS_, H_);

    // --- routing ---
    zero_i32<<<1, 256, 0, stream>>>(counts, E_);
    zero_i32<<<(SCAP + 255) / 256, 256, 0, stream>>>(assign_tok, SCAP);
    zero_i32<<<(SCAP + 255) / 256, 256, 0, stream>>>(assign_e, SCAP);
    router_k<<<T_ / 256, 256, 0, stream>>>(x, gate_w, topk_id, topk_w, counts);
    offsets_k<<<1, 32, 0, stream>>>(counts, off, fill);
    scatter_k<<<T_ / 256, 256, 0, stream>>>(topk_id, off, fill, assign_tok, assign_e, tok_pos);

    // --- shared expert: writes base result into d_out ---
    gemm1_silu<<<dim3(T_ / MT, IS_ / 256), 256, 0, stream>>>(
        xb, ws1t, act, nullptr, nullptr, nullptr, H_, IS_);
    gemm2_k<<<dim3(T_ / MT, H_ / 256), 256, 0, stream>>>(
        act, ws2t, out, nullptr, nullptr, IS_, H_);

    // --- routed experts over flat (32-aligned) assignment buckets ---
    gemm1_silu<<<dim3(SCAP / MT, I_ / 256), 256, 0, stream>>>(
        xb, w1t, act, assign_tok, assign_e, off + E_, H_, I_);
    gemm2_k<<<dim3(SCAP / MT, H_ / 256), 256, 0, stream>>>(
        act, w2t, y, assign_e, off + E_, I_, H_);

    // --- out[t] += w0*y[p0] + w1*y[p1] ---
    combine_k<<<T_, 256, 0, stream>>>(out, y, tok_pos, topk_w);
}
